// BmmEnsemble_88252987998601
// MI455X (gfx1250) — matrix operand order repaired and checked by reading, not compiled, not run on hardware
//
#include <hip/hip_runtime.h>

typedef __attribute__((ext_vector_type(16))) __bf16 v16bf;
typedef __attribute__((ext_vector_type(8)))  __bf16 v8bf;
typedef __attribute__((ext_vector_type(8)))  float  v8f;

#define NSPEC 4
#define NENS  8
#define NPER  16384          // N / NSPEC
#define NTILE 1024           // NPER / 16
#define D0    384
#define D1    160
#define D2    128
#define D3    96
#define CELU_ALPHA 0.1f

// fragments: 32 lanes x 16 bf16 = 512 elements = 1024B, fully contiguous
#define FRAG  512
#define NK0   12
#define NJ0   10
#define NK1   5
#define NJ1   8
#define NK2   4
#define NJ2   6

__device__ __forceinline__ float celu_f(float x) {
    return x > 0.f ? x : CELU_ALPHA * (__expf(x * (1.f / CELU_ALPHA)) - 1.f);
}

// A fragment (16x32 bf16) reassembled from the row-major LDS activation strip.
// Lane layout (ISA 7.12.2): lanes 0-15 row=lane, K={0..7,16..23};
// lanes 16-31 row=lane-16, K={8..15,24..31}.
__device__ __forceinline__ v16bf load_afrag_lds(const __bf16* p) {
    v8bf lo = *(const v8bf*)(p);
    v8bf hi = *(const v8bf*)(p + 16);
    v16bf a;
#pragma unroll
    for (int i = 0; i < 8; ++i) { a[i] = lo[i]; a[i + 8] = hi[i]; }
    return a;
}

// ---- prep: species-gather aev, convert fp32->bf16, and swizzle straight into
// per-wave-contiguous A fragments: xa[(tile*NK0 + k)*512 + lane*16 + t]
__global__ void gather_swizzle_kernel(const float* __restrict__ aev,
                                      const int*   __restrict__ idx,
                                      __bf16*      __restrict__ xa) {
    int gid  = blockIdx.x * blockDim.x + threadIdx.x;  // over NSPEC*NTILE*NK0*32
    int lane = gid & 31;
    int f    = gid >> 5;
    int k    = f % NK0;
    int tile = f / NK0;                                // = s*NTILE + mt
    int l15  = lane & 15;
    int hh   = lane >> 4;

    int row = idx[tile * 16 + l15];                    // idx flat [S*NPER]
    const float* base = aev + (size_t)row * D0 + k * 32 + hh * 8;
    float4 f0 = *(const float4*)(base);
    float4 f1 = *(const float4*)(base + 4);
    float4 f2 = *(const float4*)(base + 16);
    float4 f3 = *(const float4*)(base + 20);

    __bf16* o = xa + (size_t)gid * 16;
    o[0]=(__bf16)f0.x; o[1]=(__bf16)f0.y; o[2]=(__bf16)f0.z; o[3]=(__bf16)f0.w;
    o[4]=(__bf16)f1.x; o[5]=(__bf16)f1.y; o[6]=(__bf16)f1.z; o[7]=(__bf16)f1.w;
    o[8]=(__bf16)f2.x; o[9]=(__bf16)f2.y; o[10]=(__bf16)f2.z; o[11]=(__bf16)f2.w;
    o[12]=(__bf16)f3.x; o[13]=(__bf16)f3.y; o[14]=(__bf16)f3.z; o[15]=(__bf16)f3.w;
}

// ---- prep: weights [S*E][M][P] fp32 -> B-fragment swizzled bf16:
// wf[((se*NK + k)*NJ + j)*512 + lane*16 + t] holds
// w[se][ k*32 + (lane>>4)*8 + (t&7) + (t>>3)*16 ][ j*16 + (lane&15) ]
__global__ void wswizzle_kernel(const float* __restrict__ w,
                                __bf16*      __restrict__ wf,
                                int M, int P, int NJ) {
    int gid  = blockIdx.x * blockDim.x + threadIdx.x;  // over S*E*NK*NJ*32
    int lane = gid & 31;
    int f    = gid >> 5;
    int NK   = M / 32;
    int j    = f % NJ;
    int k    = (f / NJ) % NK;
    int se   = f / (NJ * NK);
    int l15  = lane & 15;
    int hh   = lane >> 4;

    const float* src = w + (size_t)se * M * P + (size_t)(k * 32) * P + (j * 16 + l15);
    __bf16* o = wf + (size_t)gid * 16;
#pragma unroll
    for (int t = 0; t < 16; ++t) o[t] = (__bf16)src[(size_t)((t & 7) + hh * 8 + (t >> 3) * 16) * P];
}

// ---- fused ensemble MLP: one wave = one (species-tile, ensemble member)
__launch_bounds__(256, 1)
__global__ void bmm_ensemble_mlp_kernel(const __bf16* __restrict__ xa,
                                        const __bf16* __restrict__ wf0,
                                        const __bf16* __restrict__ wf1,
                                        const __bf16* __restrict__ wf2,
                                        const float*  __restrict__ b0,
                                        const float*  __restrict__ b1,
                                        const float*  __restrict__ b2,
                                        const float*  __restrict__ w3,
                                        const float*  __restrict__ b3,
                                        float*        __restrict__ out) {
    __shared__ __bf16 smem[NENS * 16 * D1];            // 40 KB: 5 KB per wave

    const int e    = threadIdx.x >> 5;                 // wave id == ensemble member
    const int lane = threadIdx.x & 31;
    const int hh   = lane >> 4;
    const int l15  = lane & 15;
    const int s    = blockIdx.x >> 10;                 // NTILE == 1024
    const int mt   = blockIdx.x & (NTILE - 1);
    const int se   = s * NENS + e;

    __bf16* lds = smem + e * (16 * D1);
    const v8f vzero = {0.f, 0.f, 0.f, 0.f, 0.f, 0.f, 0.f, 0.f};

    // per-wave contiguous fragment bases (one address add per fragment after this)
    const __bf16* xaP = xa  + ((size_t)(s * NTILE + mt) * NK0) * FRAG + lane * 16;
    const __bf16* w0P = wf0 + (size_t)se * (NK0 * NJ0 * FRAG) + lane * 16;
    const __bf16* w1P = wf1 + (size_t)se * (NK1 * NJ1 * FRAG) + lane * 16;
    const __bf16* w2P = wf2 + (size_t)se * (NK2 * NJ2 * FRAG) + lane * 16;

    // ================= Layer 0: [16 x 384] @ [384 x 160] =================
    v8f c0[NJ0];
#pragma unroll
    for (int j = 0; j < NJ0; ++j) c0[j] = vzero;
#pragma unroll
    for (int k = 0; k < NK0; ++k) {
        v16bf bb[NJ0];                                  // batched: one 20-load clause
#pragma unroll
        for (int j = 0; j < NJ0; ++j)
            bb[j] = *(const v16bf*)(w0P + (size_t)(k * NJ0 + j) * FRAG);
        v16bf a = *(const v16bf*)(xaP + (size_t)k * FRAG);
#pragma unroll
        for (int j = 0; j < NJ0; ++j)
            c0[j] = __builtin_amdgcn_wmma_f32_16x16x32_bf16(
                false, a, false, bb[j], (short)0, c0[j], false, false);
    }
    // bias + CELU -> LDS [16][160] bf16 (wave-private strip; same-wave DS order)
#pragma unroll
    for (int j = 0; j < NJ0; ++j) {
        int   col  = j * 16 + l15;
        float bias = b0[se * D1 + col];
#pragma unroll
        for (int i = 0; i < 8; ++i) {
            int m = i + 8 * hh;                         // C/D layout: row = vgpr + 8*half
            lds[m * D1 + col] = (__bf16)celu_f(c0[j][i] + bias);
        }
    }

    // ================= Layer 1: [16 x 160] @ [160 x 128] =================
    v8f c1[NJ1];
#pragma unroll
    for (int j = 0; j < NJ1; ++j) c1[j] = vzero;
#pragma unroll
    for (int k = 0; k < NK1; ++k) {
        v16bf bb[NJ1];
#pragma unroll
        for (int j = 0; j < NJ1; ++j)
            bb[j] = *(const v16bf*)(w1P + (size_t)(k * NJ1 + j) * FRAG);
        v16bf a = load_afrag_lds(lds + l15 * D1 + k * 32 + hh * 8);
#pragma unroll
        for (int j = 0; j < NJ1; ++j)
            c1[j] = __builtin_amdgcn_wmma_f32_16x16x32_bf16(
                false, a, false, bb[j], (short)0, c1[j], false, false);
    }
#pragma unroll
    for (int j = 0; j < NJ1; ++j) {
        int   col  = j * 16 + l15;
        float bias = b1[se * D2 + col];
#pragma unroll
        for (int i = 0; i < 8; ++i) {
            int m = i + 8 * hh;
            lds[m * D2 + col] = (__bf16)celu_f(c1[j][i] + bias);
        }
    }

    // ================= Layer 2: [16 x 128] @ [128 x 96] ==================
    v8f c2[NJ2];
#pragma unroll
    for (int j = 0; j < NJ2; ++j) c2[j] = vzero;
#pragma unroll
    for (int k = 0; k < NK2; ++k) {
        v16bf bb[NJ2];
#pragma unroll
        for (int j = 0; j < NJ2; ++j)
            bb[j] = *(const v16bf*)(w2P + (size_t)(k * NJ2 + j) * FRAG);
        v16bf a = load_afrag_lds(lds + l15 * D2 + k * 32 + hh * 8);
#pragma unroll
        for (int j = 0; j < NJ2; ++j)
            c2[j] = __builtin_amdgcn_wmma_f32_16x16x32_bf16(
                false, a, false, bb[j], (short)0, c2[j], false, false);
    }

    // ====== Layer 3 + ensemble mean + global sum, folded analytically =====
    float acc = 0.f;
#pragma unroll
    for (int j = 0; j < NJ2; ++j) {
        int   col  = j * 16 + l15;
        float bias = b2[se * D3 + col];
        float cs   = 0.f;
#pragma unroll
        for (int i = 0; i < 8; ++i) cs += celu_f(c2[j][i] + bias);
        acc += cs * w3[se * D3 + col];
    }
#pragma unroll
    for (int off = 16; off >= 1; off >>= 1) acc += __shfl_xor(acc, off, 32);

    if (lane == 0) {
        float val = (acc + 16.f * b3[se]) * (1.0f / NENS);  // 16 atoms, mean over E
        atomicAdd(out, val);
    }
}

extern "C" void kernel_launch(void* const* d_in, const int* in_sizes, int n_in,
                              void* d_out, int out_size, void* d_ws, size_t ws_size,
                              hipStream_t stream) {
    const float* aev = (const float*)d_in[0];
    // d_in[1] = species (unused: idx fully encodes the species gather)
    const int*   idx = (const int*)d_in[2];
    const float* w0  = (const float*)d_in[3];
    const float* b0  = (const float*)d_in[4];
    const float* w1  = (const float*)d_in[5];
    const float* b1  = (const float*)d_in[6];
    const float* w2  = (const float*)d_in[7];
    const float* b2  = (const float*)d_in[8];
    const float* w3  = (const float*)d_in[9];
    const float* b3  = (const float*)d_in[10];
    float* out = (float*)d_out;

    // workspace (bytes, all 1KB aligned):
    //   xa  : 4096 tiles * 12 frags * 1KB = 50,331,648
    //   wf0 : 32 se * 120 frags * 1KB    =  3,932,160
    //   wf1 : 32 se *  40 frags * 1KB    =  1,310,720
    //   wf2 : 32 se *  24 frags * 1KB    =    786,432
    char* ws = (char*)d_ws;
    __bf16* xa  = (__bf16*)(ws);
    __bf16* wf0 = (__bf16*)(ws + 50331648u);
    __bf16* wf1 = (__bf16*)(ws + 54263808u);
    __bf16* wf2 = (__bf16*)(ws + 55574528u);

    hipMemsetAsync(d_out, 0, sizeof(float) * (size_t)out_size, stream);

    {   int total = NSPEC * NTILE * NK0 * 32;          // 1,572,864
        gather_swizzle_kernel<<<total / 256, 256, 0, stream>>>(aev, idx, xa); }
    {   int total = NSPEC * NENS * NK0 * NJ0 * 32;     // 122,880
        wswizzle_kernel<<<total / 256, 256, 0, stream>>>(w0, wf0, D0, D1, NJ0); }
    {   int total = NSPEC * NENS * NK1 * NJ1 * 32;     // 40,960
        wswizzle_kernel<<<total / 256, 256, 0, stream>>>(w1, wf1, D1, D2, NJ1); }
    {   int total = NSPEC * NENS * NK2 * NJ2 * 32;     // 24,576
        wswizzle_kernel<<<total / 256, 256, 0, stream>>>(w2, wf2, D2, D3, NJ2); }

    // 4 species * 1024 atom-tiles of 16; 8 waves/block = 8 ensemble members
    bmm_ensemble_mlp_kernel<<<NSPEC * NTILE, 256, 0, stream>>>(
        xa, wf0, wf1, wf2, b0, b1, b2, w3, b3, out);
}
